// ParcaeCausalSelfAttention_19361712570795
// MI455X (gfx1250) — matrix operand order repaired and checked by reading, not compiled, not run on hardware
//
#include <hip/hip_runtime.h>

// ---------------------------------------------------------------------------
// ParcaeCausalSelfAttention for MI455X (gfx1250, wave32, WMMA bf16)
// Pipeline: f32->bf16 convert -> 3x WMMA GEMM (QKV, async double-buffered
// LDS staging) -> epilogue (clip/bias/ve-gate/RoPE/RMSnorm) -> flash
// attention (WMMA + async staging) -> WMMA GEMM output projection.
// ---------------------------------------------------------------------------

typedef unsigned short u16;
typedef unsigned int   u32;
typedef __attribute__((ext_vector_type(16))) __bf16 v16bf;
typedef __attribute__((ext_vector_type(8)))  __bf16 v8bf;
typedef __attribute__((ext_vector_type(8)))  float  v8f;
typedef int v4i_gcc __attribute__((vector_size(16)));   // matches builtin param

#define BATCH 2
#define SEQ   2048
#define DIM   2048
#define NH    16
#define HD    128
#define RD    64
#define VECH  32
#define MROWS (BATCH * SEQ)   // 4096
#define CLIPV 8.0f
#define EPSV  1e-6f

#define AS1 __attribute__((address_space(1)))
#define AS3 __attribute__((address_space(3)))

// --- async global->LDS staging (CDNA5 GLOBAL_LOAD_ASYNC_TO_LDS_B128) -------
#if __has_builtin(__builtin_amdgcn_global_load_async_to_lds_b128) && \
    __has_builtin(__builtin_amdgcn_s_wait_asynccnt)
#define CP16(lds, gbl)                                                       \
  __builtin_amdgcn_global_load_async_to_lds_b128((AS1 v4i_gcc*)(gbl),        \
                                                 (AS3 v4i_gcc*)(lds), 0, 0)
#define WAIT_ASYNC(n) __builtin_amdgcn_s_wait_asynccnt(n)
#else
#define CP16(lds, gbl) (*(uint4*)(lds) = *(const uint4*)(gbl))
#define WAIT_ASYNC(n)
#endif

union AFrag { v16bf v; v8bf h[2]; };

__device__ __forceinline__ u16 f2bf(float f) {
  u32 u = __float_as_uint(f);
  u32 r = u + 0x7FFFu + ((u >> 16) & 1u);   // round-to-nearest-even
  return (u16)(r >> 16);
}

// ---------------------------------------------------------------------------
// f32 -> bf16 conversion (4 elems / thread)
// ---------------------------------------------------------------------------
__global__ __launch_bounds__(256) void cvt_bf16(const float* __restrict__ in,
                                                u16* __restrict__ out, int n) {
  int i = (blockIdx.x * 256 + threadIdx.x) * 4;
  if (i + 3 < n) {
    float4 f = *(const float4*)(in + i);
    out[i + 0] = f2bf(f.x);
    out[i + 1] = f2bf(f.y);
    out[i + 2] = f2bf(f.z);
    out[i + 3] = f2bf(f.w);
  }
}

// ---------------------------------------------------------------------------
// C[M,N] (f32) = A[M,K] (bf16 rowmajor) * W[N,K]^T (bf16 rowmajor)
// Block tile 128x128x64, double-buffered async LDS staging, peeled drain.
// 8 waves, each wave 64x32 (4x2 wmma frags), 16 WMMA per wave per tile.
// ---------------------------------------------------------------------------
#define TM 128
#define TN 128
#define TK 64

__global__ __launch_bounds__(256)
void gemm_bf16(const u16* __restrict__ A, const u16* __restrict__ W,
               float* __restrict__ C, int M, int N, int K) {
  __shared__ u16 As[2][TM][TK];   // 2 x 16 KB
  __shared__ u16 Ws[2][TN][TK];   // 2 x 16 KB

  const int bm = blockIdx.y * TM, bn = blockIdx.x * TN;
  const int tid  = threadIdx.x;
  const int lane = tid & 31, wid = tid >> 5;
  const int wm = (wid >> 2) * 64;   // wave M origin in tile
  const int wn = (wid & 3) * 32;    // wave N origin in tile
  const int lh = lane & 15, lg = lane >> 4;

  v8f acc[4][2] = {};

  // staging slot: each thread moves 64B of A and 64B of W per tile
  const int ar = tid >> 1;            // 0..127
  const int ak = (tid & 1) * 32;      // 0 or 32 (elems)
  const u16* gA = A + (size_t)(bm + ar) * K + ak;
  const u16* gW = W + (size_t)(bn + ar) * K + ak;

  auto stage = [&](int buf, int k0) {
    CP16(&As[buf][ar][ak +  0], gA + k0 +  0);
    CP16(&As[buf][ar][ak +  8], gA + k0 +  8);
    CP16(&As[buf][ar][ak + 16], gA + k0 + 16);
    CP16(&As[buf][ar][ak + 24], gA + k0 + 24);
    CP16(&Ws[buf][ar][ak +  0], gW + k0 +  0);
    CP16(&Ws[buf][ar][ak +  8], gW + k0 +  8);
    CP16(&Ws[buf][ar][ak + 16], gW + k0 + 16);
    CP16(&Ws[buf][ar][ak + 24], gW + k0 + 24);
  };

  auto compute = [&](int buf) {
#pragma unroll
    for (int kk = 0; kk < TK; kk += 32) {
      v16bf bfrag[2];
#pragma unroll
      for (int ni = 0; ni < 2; ni++)
      { AFrag bu;
        bu.h[0] = *(const v8bf*)&Ws[buf][wn + ni * 16 + lh][kk + lg * 8];
        bu.h[1] = *(const v8bf*)&Ws[buf][wn + ni * 16 + lh][kk + 16 + lg * 8];
        bfrag[ni] = bu.v; }
#pragma unroll
      for (int mi = 0; mi < 4; mi++) {
        AFrag af;
        af.h[0] = *(const v8bf*)&As[buf][wm + mi * 16 + lh][kk + lg * 8];
        af.h[1] = *(const v8bf*)&As[buf][wm + mi * 16 + lh][kk + 16 + lg * 8];
#pragma unroll
        for (int ni = 0; ni < 2; ni++)
          acc[mi][ni] = __builtin_amdgcn_wmma_f32_16x16x32_bf16(
              false, af.v, false, bfrag[ni], (short)0, acc[mi][ni], false, false);
      }
    }
  };

  const int ntiles = K / TK;
  stage(0, 0);
  int kt = 0;
  for (; kt < ntiles - 1; kt++) {     // steady state: unconditional wait(8)
    stage((kt & 1) ^ 1, (kt + 1) * TK);
    WAIT_ASYNC(8);
    __syncthreads();
    compute(kt & 1);
    __syncthreads();
  }
  WAIT_ASYNC(0);                      // drain: last tile
  __syncthreads();
  compute(kt & 1);

#pragma unroll
  for (int mi = 0; mi < 4; mi++)
#pragma unroll
    for (int ni = 0; ni < 2; ni++)
#pragma unroll
      for (int r = 0; r < 8; r++) {
        int row = bm + wm + mi * 16 + r + lg * 8;
        int col = bn + wn + ni * 16 + lh;
        C[(size_t)row * N + col] = acc[mi][ni][r];
      }
}

// ---------------------------------------------------------------------------
// QKV epilogue. One wave per (token m, head h): clip, bias, ve-gate, RoPE,
// RMS-norm. Lane owns 4 consecutive elems (RoPE pairs stay in-lane).
// Outputs: qb/kb bf16 [m][h*128+d]; v transposed bf16 [ (b*NH+h)*HD+d ][ s ].
// ---------------------------------------------------------------------------
__global__ __launch_bounds__(256)
void qkv_epilogue(const float* __restrict__ qf, const float* __restrict__ kf,
                  const float* __restrict__ vf, const float* __restrict__ x,
                  const float* __restrict__ ve, const float* __restrict__ freqs,
                  const float* __restrict__ qk_bias,
                  const float* __restrict__ ve_gate_w,
                  u16* __restrict__ qb, u16* __restrict__ kb,
                  u16* __restrict__ vbt) {
  const int lane = threadIdx.x & 31, wid = threadIdx.x >> 5;
  const int pid = blockIdx.x * 8 + wid;       // (m,h) pair id
  const int m = pid >> 4, h = pid & 15;
  const int b = m >> 11, s = m & 2047;
  const size_t base = (size_t)m * DIM + h * HD;

  float q[4], k[4], v[4];
  const float4 q4 = ((const float4*)(qf + base))[lane];
  const float4 k4 = ((const float4*)(kf + base))[lane];
  const float4 v4 = ((const float4*)(vf + base))[lane];
  q[0] = q4.x; q[1] = q4.y; q[2] = q4.z; q[3] = q4.w;
  k[0] = k4.x; k[1] = k4.y; k[2] = k4.z; k[3] = k4.w;
  v[0] = v4.x; v[1] = v4.y; v[2] = v4.z; v[3] = v4.w;

  // ve gate: 2*sigmoid( x[m, 0:32] . ve_gate_w[h, :] ), one channel per lane
  float g = x[(size_t)m * DIM + lane] * ve_gate_w[h * VECH + lane];
#pragma unroll
  for (int o = 16; o; o >>= 1) g += __shfl_xor(g, o, 32);
  const float gate = 2.0f / (1.0f + __expf(-g));

  const float4 ve4 = ((const float4*)(ve + base))[lane];
  v[0] += gate * ve4.x; v[1] += gate * ve4.y;
  v[2] += gate * ve4.z; v[3] += gate * ve4.w;

#pragma unroll
  for (int j = 0; j < 4; j++) {
    int e = lane * 4 + j;
    q[j] = fminf(fmaxf(q[j], -CLIPV), CLIPV) + qk_bias[0 * NH * HD + h * HD + e];
    k[j] = fminf(fmaxf(k[j], -CLIPV), CLIPV) + qk_bias[1 * NH * HD + h * HD + e];
    v[j] = fminf(fmaxf(v[j], -CLIPV), CLIPV);
  }

  // RoPE on first RD=64 elems (lanes 0..15): pairs (2l, 2l+1)
  if (lane < 16) {
    const float c0 = freqs[s * 128 + (2 * lane) * 2];
    const float s0 = freqs[s * 128 + (2 * lane) * 2 + 1];
    const float c1 = freqs[s * 128 + (2 * lane + 1) * 2];
    const float s1 = freqs[s * 128 + (2 * lane + 1) * 2 + 1];
    float t;
    t = q[0] * c0 - q[1] * s0; q[1] = q[1] * c0 + q[0] * s0; q[0] = t;
    t = q[2] * c1 - q[3] * s1; q[3] = q[3] * c1 + q[2] * s1; q[2] = t;
    t = k[0] * c0 - k[1] * s0; k[1] = k[1] * c0 + k[0] * s0; k[0] = t;
    t = k[2] * c1 - k[3] * s1; k[3] = k[3] * c1 + k[2] * s1; k[2] = t;
  }

  // RMS norm over the 128-elem head row
  float sq = q[0]*q[0] + q[1]*q[1] + q[2]*q[2] + q[3]*q[3];
  float sk = k[0]*k[0] + k[1]*k[1] + k[2]*k[2] + k[3]*k[3];
#pragma unroll
  for (int o = 16; o; o >>= 1) { sq += __shfl_xor(sq, o, 32); sk += __shfl_xor(sk, o, 32); }
  const float rq = rsqrtf(sq * (1.0f / HD) + EPSV);
  const float rk = rsqrtf(sk * (1.0f / HD) + EPSV);

#pragma unroll
  for (int j = 0; j < 4; j++) {
    const int e = lane * 4 + j;
    qb[base + e] = f2bf(q[j] * rq);
    kb[base + e] = f2bf(k[j] * rk);
    vbt[((size_t)((b * NH + h) * HD + e)) * SEQ + s] = f2bf(v[j]);
  }
}

// ---------------------------------------------------------------------------
// Flash attention. Block = (128 q rows, head h, batch b); 8 waves, each wave
// owns 16 q rows. K/V^T tiles of 32 keys double-buffered in LDS via async
// copies (peeled drain). Online softmax with row stats kept lane-wise in the
// native 16x16 C layout; P converted to A-frag layout via a per-wave LDS pad.
// ---------------------------------------------------------------------------
#define QT 128
#define KT 32

__global__ __launch_bounds__(256)
void flash_attn(const u16* __restrict__ qb, const u16* __restrict__ kb,
                const u16* __restrict__ vbt, u16* __restrict__ yb) {
  __shared__ u16 Ks[2][KT][HD];    // [key][hd]  2 x 8 KB
  __shared__ u16 Vs[2][HD][KT];    // [hd][key]  2 x 8 KB (V^T)
  __shared__ u16 Ps[8][16][KT];    // per-wave P pad, 8 KB

  const int q0 = blockIdx.x * QT;
  const int h = blockIdx.y, b = blockIdx.z;
  const int tid = threadIdx.x, lane = tid & 31, wid = tid >> 5;
  const int lh = lane & 15, lg = lane >> 4;

  // --- Q fragments resident in registers for the whole KV loop ---
  const int qrow = q0 + wid * 16 + lh;                // A-row for this lane
  const size_t qg = ((size_t)(b * SEQ + qrow)) * DIM + h * HD;
  v16bf qfrag[4];
#pragma unroll
  for (int t = 0; t < 4; t++) {
    AFrag u;
    u.h[0] = *(const v8bf*)(qb + qg + t * 32 + lg * 8);
    u.h[1] = *(const v8bf*)(qb + qg + t * 32 + 16 + lg * 8);
    qfrag[t] = u.v;
  }

  v8f o[8] = {};                 // O: 16x128, 8 N-frags
  float mrow[8], lrow[8];
#pragma unroll
  for (int r = 0; r < 8; r++) { mrow[r] = -1.0e30f; lrow[r] = 0.0f; }

  // staging slots: 32B of K + 32B of V^T per thread per tile
  const int kr = tid >> 3, kc = (tid & 7) * 16;
  const int vr = tid >> 1, vc = (tid & 1) * 16;
  const u16* gK = kb + ((size_t)(b * SEQ + kr)) * DIM + h * HD + kc;
  const u16* gV = vbt + ((size_t)((b * NH + h) * HD + vr)) * SEQ + vc;

  auto stage = [&](int buf, int k0) {
    CP16(&Ks[buf][kr][kc],     gK + (size_t)k0 * DIM);
    CP16(&Ks[buf][kr][kc + 8], gK + (size_t)k0 * DIM + 8);
    CP16(&Vs[buf][vr][vc],     gV + k0);
    CP16(&Vs[buf][vr][vc + 8], gV + k0 + 8);
  };

  const float scale = 0.08838834764831845f;   // 1/sqrt(128)

  auto tile_compute = [&](int kt) {
    const int k0 = kt * KT;
    const int buf = kt & 1;

    // S = Q K^T  (two 16x16 C frags: keys [0,16) and [16,32))
    v8f sf[2] = {};
#pragma unroll
    for (int g = 0; g < 2; g++)
#pragma unroll
      for (int t = 0; t < 4; t++) {
        AFrag bu;
        bu.h[0] = *(const v8bf*)&Ks[buf][g * 16 + lh][t * 32 + lg * 8];
        bu.h[1] = *(const v8bf*)&Ks[buf][g * 16 + lh][t * 32 + 16 + lg * 8];
        v16bf bf = bu.v;
        sf[g] = __builtin_amdgcn_wmma_f32_16x16x32_bf16(
            false, qfrag[t], false, bf, (short)0, sf[g], false, false);
      }

    // online softmax update (per C-layout row r, true row = r + 8*lg)
    const int sqbase = q0 + wid * 16 + lg * 8;
#pragma unroll
    for (int r = 0; r < 8; r++) {
      const int row_s = sqbase + r;
      float e0 = sf[0][r] * scale;
      float e1 = sf[1][r] * scale;
      if (k0 + lh > row_s)      e0 = -3.0e38f;   // causal mask
      if (k0 + 16 + lh > row_s) e1 = -3.0e38f;
      float mx = fmaxf(e0, e1);
#pragma unroll
      for (int off = 8; off; off >>= 1) mx = fmaxf(mx, __shfl_xor(mx, off, 16));
      const float mn = fmaxf(fmaxf(mrow[r], mx), -1.0e30f);
      const float corr = __expf(mrow[r] - mn);
      const float p0 = __expf(e0 - mn);
      const float p1 = __expf(e1 - mn);
      float rs = p0 + p1;
#pragma unroll
      for (int off = 8; off; off >>= 1) rs += __shfl_xor(rs, off, 16);
      lrow[r] = lrow[r] * corr + rs;
      mrow[r] = mn;
#pragma unroll
      for (int n = 0; n < 8; n++) o[n][r] *= corr;
      Ps[wid][r + 8 * lg][lh]      = f2bf(p0);
      Ps[wid][r + 8 * lg][16 + lh] = f2bf(p1);
    }

    // P (A-frag layout from per-wave pad), then O += P @ V
    AFrag pu;
    pu.h[0] = *(const v8bf*)&Ps[wid][lh][lg * 8];
    pu.h[1] = *(const v8bf*)&Ps[wid][lh][16 + lg * 8];
#pragma unroll
    for (int n = 0; n < 8; n++) {
      AFrag vu;
      vu.h[0] = *(const v8bf*)&Vs[buf][n * 16 + lh][lg * 8];
      vu.h[1] = *(const v8bf*)&Vs[buf][n * 16 + lh][16 + lg * 8];
      v16bf bv = vu.v;
      o[n] = __builtin_amdgcn_wmma_f32_16x16x32_bf16(
          false, pu.v, false, bv, (short)0, o[n], false, false);
    }
  };

  const int nk = (q0 + QT) / KT;   // causal span of this q tile (>= 4)
  stage(0, 0);
  int kt = 0;
  for (; kt < nk - 1; kt++) {      // steady state: unconditional wait(4)
    stage((kt & 1) ^ 1, (kt + 1) * KT);
    WAIT_ASYNC(4);
    __syncthreads();
    tile_compute(kt);
    __syncthreads();
  }
  WAIT_ASYNC(0);                   // drain: last tile
  __syncthreads();
  tile_compute(kt);

  // normalize and store (bf16 for the final projection GEMM)
#pragma unroll
  for (int r = 0; r < 8; r++) {
    const float inv = 1.0f / lrow[r];
    const int srow = q0 + wid * 16 + r + lg * 8;
    const size_t ob = ((size_t)(b * SEQ + srow)) * DIM + h * HD;
#pragma unroll
    for (int n = 0; n < 8; n++)
      yb[ob + n * 16 + lh] = f2bf(o[n][r] * inv);
  }
}

// ---------------------------------------------------------------------------
// Host-side orchestration
// ---------------------------------------------------------------------------
extern "C" void kernel_launch(void* const* d_in, const int* in_sizes, int n_in,
                              void* d_out, int out_size, void* d_ws, size_t ws_size,
                              hipStream_t stream) {
  const float* x    = (const float*)d_in[0];
  const float* fcis = (const float*)d_in[1];
  const float* ve   = (const float*)d_in[2];
  const float* Wq   = (const float*)d_in[3];
  const float* Wk   = (const float*)d_in[4];
  const float* Wv   = (const float*)d_in[5];
  const float* Wo   = (const float*)d_in[6];
  const float* qkb  = (const float*)d_in[7];
  const float* vgw  = (const float*)d_in[8];
  float* out = (float*)d_out;
  char*  ws  = (char*)d_ws;

  const size_t MB = 1024ull * 1024ull;
  u16*   wq_b = (u16*)(ws + 0 * MB);      //  8 MiB
  u16*   wk_b = (u16*)(ws + 8 * MB);      //  8 MiB
  u16*   wv_b = (u16*)(ws + 16 * MB);     //  8 MiB
  u16*   wo_b = (u16*)(ws + 24 * MB);     //  8 MiB
  u16*   xb   = (u16*)(ws + 32 * MB);     // 16 MiB  (x bf16; later reused as qb)
  float* qf   = (float*)(ws + 48 * MB);   // 32 MiB  (later reused as yb)
  float* kf   = (float*)(ws + 80 * MB);   // 32 MiB
  float* vf   = (float*)(ws + 112 * MB);  // 32 MiB
  u16*   kb   = (u16*)(ws + 144 * MB);    // 16 MiB
  u16*   vbt  = (u16*)(ws + 160 * MB);    // 16 MiB   -> 176 MiB total
  u16*   qb = xb;                          // alias: xb dead after QKV GEMMs
  u16*   yb = (u16*)qf;                    // alias: qf dead after epilogue

  const int NW = DIM * DIM;                // 4194304 weight elems
  const int NX = MROWS * DIM;              // 8388608 activation elems

  // 1) downconvert
  cvt_bf16<<<NX / 1024, 256, 0, stream>>>(x,  xb,   NX);
  cvt_bf16<<<NW / 1024, 256, 0, stream>>>(Wq, wq_b, NW);
  cvt_bf16<<<NW / 1024, 256, 0, stream>>>(Wk, wk_b, NW);
  cvt_bf16<<<NW / 1024, 256, 0, stream>>>(Wv, wv_b, NW);
  cvt_bf16<<<NW / 1024, 256, 0, stream>>>(Wo, wo_b, NW);

  // 2) QKV projections (WMMA)
  dim3 ggrid(DIM / TN, MROWS / TM);        // (16, 32)
  gemm_bf16<<<ggrid, 256, 0, stream>>>(xb, wq_b, qf, MROWS, DIM, DIM);
  gemm_bf16<<<ggrid, 256, 0, stream>>>(xb, wk_b, kf, MROWS, DIM, DIM);
  gemm_bf16<<<ggrid, 256, 0, stream>>>(xb, wv_b, vf, MROWS, DIM, DIM);

  // 3) epilogue: clip/bias/gate/RoPE/RMSnorm, bf16 repack (v transposed)
  qkv_epilogue<<<(MROWS * NH) / 8, 256, 0, stream>>>(
      qf, kf, vf, x, ve, fcis, qkb, vgw, qb, kb, vbt);

  // 4) flash attention (WMMA)
  flash_attn<<<dim3(SEQ / QT, NH, BATCH), 256, 0, stream>>>(qb, kb, vbt, yb);

  // 5) output projection (WMMA) straight into d_out (f32)
  gemm_bf16<<<ggrid, 256, 0, stream>>>(yb, wo_b, out, MROWS, DIM, DIM);
}
